// Attention_25786983645615
// MI455X (gfx1250) — matrix operand order repaired and checked by reading, not compiled, not run on hardware
//
#include <hip/hip_runtime.h>
#include <hip/hip_bf16.h>

// ---------------------------------------------------------------------------
// Problem constants (fixed by reference setup_inputs):
//   B=8, T=40 (text), S=1024 (image), N=T+S=1064, C=768, H=12, hd=64
// ---------------------------------------------------------------------------
#define M_TOK   8512
#define C_DIM   768
#define QKV_DIM 2304
#define N_SEQ   1064
#define T_TXT   40
#define H_NUM   12

typedef __attribute__((ext_vector_type(16))) __bf16 bf16x16;
typedef __attribute__((ext_vector_type(8)))  __bf16 bf16x8;
typedef __attribute__((ext_vector_type(8)))  float  v8f;

__device__ __forceinline__ __bf16 to_bf16(float f) { return (__bf16)f; }

__device__ __forceinline__ v8f wmma_bf16(bf16x16 a, bf16x16 b, v8f c) {
  return __builtin_amdgcn_wmma_f32_16x16x32_bf16(false, a, false, b, (short)0, c,
                                                 false, false);
}

__device__ __forceinline__ bf16x16 cat8(bf16x8 lo, bf16x8 hi) {
  return __builtin_shufflevector(lo, hi, 0, 1, 2, 3, 4, 5, 6, 7,
                                 8, 9, 10, 11, 12, 13, 14, 15);
}

// Issue one per-lane 16B async copy global -> LDS (ASYNCcnt-tracked)
__device__ __forceinline__ void async_g2l_b128(unsigned lds_off,
                                               const __bf16* gsrc) {
  asm volatile("global_load_async_to_lds_b128 %0, %1, off"
               :: "v"(lds_off), "v"(gsrc) : "memory");
}
__device__ __forceinline__ void wait_async0() {
  asm volatile("s_wait_asynccnt 0x0" ::: "memory");
}

// A-fragment (16x32, 16-bit): lane holds row (lane&15); element e holds
// K = e + 16*(lane>=16). Two aligned 16B loads.
__device__ __forceinline__ bf16x16 load_afrag(const __bf16* __restrict__ base,
                                              size_t row0, int ld, int k0) {
  const int lane = threadIdx.x & 31;
  const __bf16* p = base + (row0 + (size_t)(lane & 15)) * (size_t)ld + k0 +
                    ((lane >> 4) << 4);
  return cat8(*(const bf16x8*)p, *(const bf16x8*)(p + 8));
}

// Guarded B-fragment for K-matrix tiles (column = key token, zero beyond limit)
__device__ __forceinline__ bf16x16 load_kfrag(const __bf16* __restrict__ K,
                                              int key0, int k0, int ld,
                                              int limit) {
  const int lane = threadIdx.x & 31;
  const int key = key0 + (lane & 15);
  bf16x16 r = {};
  if (key < limit) {
    const __bf16* p = K + (size_t)key * (size_t)ld + k0 + ((lane >> 4) << 4);
    r = cat8(*(const bf16x8*)p, *(const bf16x8*)(p + 8));
  }
  return r;
}

// 16-lane XOR butterfly via v_permlane16_b32 (backend-verified on gfx1250)
__device__ __forceinline__ float lanexor16(float v, unsigned s0, unsigned s1) {
  unsigned u = __builtin_bit_cast(unsigned, v);
  u = __builtin_amdgcn_permlane16(u, u, s0, s1, false, false);
  return __builtin_bit_cast(float, u);
}
__device__ __forceinline__ float rowmax16(float v) {
  v = fmaxf(v, lanexor16(v, 0x67452301u, 0xEFCDAB89u));  // ^1
  v = fmaxf(v, lanexor16(v, 0x45670123u, 0xCDEF89ABu));  // ^2
  v = fmaxf(v, lanexor16(v, 0x01234567u, 0x89ABCDEFu));  // ^4
  v = fmaxf(v, lanexor16(v, 0xFEDCBA98u, 0x76543210u));  // ^8
  return v;
}

// ---------------------------------------------------------------------------
// f32 -> bf16 conversion (one element per thread)
// ---------------------------------------------------------------------------
__global__ void __launch_bounds__(256, 1)
f32_to_bf16(const float* __restrict__ src, __bf16* __restrict__ dst, int n) {
  int i = blockIdx.x * blockDim.x + threadIdx.x;
  if (i < n) dst[i] = to_bf16(src[i]);
}

// ---------------------------------------------------------------------------
// WMMA GEMM: C[M,N] = A[M,K] * W[N,K]^T (+ bias).
// Block = 8 waves x 32 rows = 256 M-rows x 64 N-cols. The shared 64x32 W tile
// is staged into LDS with GLOBAL_LOAD_ASYNC_TO_LDS_B128, *double-buffered*:
// the K-step k+1 tile is prefetched right after the barrier and the DMA runs
// concurrently with the 8 WMMAs of step k. One s_wait_asynccnt + one barrier
// per K-step.
// ---------------------------------------------------------------------------
#define WSTR 40  // padded LDS row stride (elements) to spread banks
template <bool OUT_BF16>
__global__ void __launch_bounds__(256, 1)
gemm_wmma(const __bf16* __restrict__ A, const __bf16* __restrict__ W,
          __bf16* __restrict__ Ob, float* __restrict__ Of,
          const float* __restrict__ bias, int M, int N, int K) {
  __shared__ __bf16 wt[2][64 * WSTR];  // double-buffered W tile
  const int lane = threadIdx.x & 31;
  const int wib = threadIdx.x >> 5;
  const int ngroups = N >> 6;
  const int mtb = blockIdx.x / ngroups;
  const int ng = blockIdx.x % ngroups;
  const int ncol0 = ng * 64;
  size_t row0 = (size_t)mtb * 256 + (size_t)wib * 32;
  if (row0 + 32 > (size_t)M) row0 = (size_t)M - 32;  // tail clamp (dup work)

  // async-staging assignment: 256 threads x 16B cover the 64x32 bf16 W tile
  const int trow = threadIdx.x >> 2;         // W row (output column) 0..63
  const int tseg = (threadIdx.x & 3) << 3;   // K segment 0,8,16,24
  const __bf16* wsrc = W + (size_t)(ncol0 + trow) * K + tseg;
  const unsigned ldsoff0 = (unsigned)(size_t)(&wt[0][0] + trow * WSTR + tseg);
  const unsigned ldsoff1 = (unsigned)(size_t)(&wt[1][0] + trow * WSTR + tseg);

  // prologue: prefetch first tile into buffer 0
  async_g2l_b128(ldsoff0, wsrc);

  v8f acc[2][4] = {};
  int cur = 0;
#pragma unroll 1
  for (int k0 = 0; k0 < K; k0 += 32) {
    wait_async0();      // own prefetch (tile k0) complete
    __syncthreads();    // all waves: prefetch done + previous reads retired

    if (k0 + 32 < K)    // prefetch next tile into the other buffer (overlaps)
      async_g2l_b128(cur ? ldsoff0 : ldsoff1, wsrc + k0 + 32);

    bf16x16 a0 = load_afrag(A, row0, K, k0);
    bf16x16 a1 = load_afrag(A, row0 + 16, K, k0);
    const __bf16* wb = &wt[cur][0];
#pragma unroll
    for (int j = 0; j < 4; ++j) {
      const __bf16* wp = wb + (j * 16 + (lane & 15)) * WSTR + ((lane >> 4) << 4);
      bf16x16 b = cat8(*(const bf16x8*)wp, *(const bf16x8*)(wp + 8));
      acc[0][j] = wmma_bf16(a0, b, acc[0][j]);
      acc[1][j] = wmma_bf16(a1, b, acc[1][j]);
    }
    cur ^= 1;
  }

  const int col = lane & 15;
  const int rh = (lane >> 4) << 3;  // +8 rows for upper half-wave
#pragma unroll
  for (int j = 0; j < 4; ++j) {
    const int c = ncol0 + j * 16 + col;
    const float bv = OUT_BF16 ? 0.f : bias[c];
#pragma unroll
    for (int half = 0; half < 2; ++half) {
#pragma unroll
      for (int r = 0; r < 8; ++r) {
        const size_t idx = (row0 + half * 16 + r + rh) * (size_t)N + c;
        if (OUT_BF16)
          Ob[idx] = to_bf16(acc[half][j][r]);
        else
          Of[idx] = acc[half][j][r] + bv;
      }
    }
  }
}

// ---------------------------------------------------------------------------
// Flash-style attention. qkv: (8512 x 2304) bf16; out: (8512 x 768) bf16.
// One wave per 16-query tile of one (b,h):
//   tiles 0..2  -> text queries 0..39  (keys limited to 40)
//   tiles 3..66 -> image queries 40..1063 (all 1064 keys)
// Per 32-key step: 4 score WMMAs + 1 row-sum WMMA (P x ones) + 4 P*V WMMAs.
// Softmax row-max via v_permlane16 butterflies; V tile staged to LDS
// transposed; P tile staged as bf16 (b128 A-fragment reloads).
// ---------------------------------------------------------------------------
#define ATTN_WAVES 4
#define PSTR 40
__global__ void __launch_bounds__(32 * ATTN_WAVES, 1)
attn_wmma(const __bf16* __restrict__ qkv, __bf16* __restrict__ out) {
  __shared__ __bf16 s_p[ATTN_WAVES][16 * PSTR];  // P tile:   [row][key]
  __shared__ __bf16 s_v[ATTN_WAVES][64 * PSTR];  // V tile^T: [hdcol][key]
  const int lane = threadIdx.x & 31;
  const int wib = threadIdx.x >> 5;
  const int wave = blockIdx.x * ATTN_WAVES + wib;
  const int qt = wave % 67;
  const int bh = wave / 67;
  if (bh >= 8 * H_NUM) return;
  const int h = bh % H_NUM;
  const int b = bh / H_NUM;

  int q0, limit_q, limit_k;
  if (qt < 3) {
    q0 = qt * 16; limit_q = T_TXT; limit_k = T_TXT;
  } else {
    q0 = T_TXT + (qt - 3) * 16; limit_q = N_SEQ; limit_k = N_SEQ;
  }

  const size_t tokbase = (size_t)b * N_SEQ;
  const __bf16* qb = qkv + tokbase * QKV_DIM + h * 64;
  const __bf16* kb_ = qkv + tokbase * QKV_DIM + C_DIM + h * 64;
  const __bf16* vb = qkv + tokbase * QKV_DIM + 2 * C_DIM + h * 64;

  const bf16x16 qa0 = load_afrag(qb, (size_t)q0, QKV_DIM, 0);
  const bf16x16 qa1 = load_afrag(qb, (size_t)q0, QKV_DIM, 32);

  bf16x16 ones;
#pragma unroll
  for (int e = 0; e < 16; ++e) ones[e] = to_bf16(1.0f);

  v8f acc[4] = {};
  float m[8], l[8];
#pragma unroll
  for (int r = 0; r < 8; ++r) {
    m[r] = -__builtin_inff();
    l[r] = 0.f;
  }

  __bf16* sp = s_p[wib];
  __bf16* sv = s_v[wib];
  const int col = lane & 15;
  const int rh = (lane >> 4) << 3;
  const int c8 = (lane & 7) * 8;   // hd col base for V staging
  const int kl4 = lane >> 3;       // key sub-index for V staging
  const float scale = 0.125f;      // hd^-0.5

#pragma unroll 1
  for (int kb = 0; kb < limit_k; kb += 32) {
    // ---- stage V tile (32 keys x 64 hd) into LDS transposed, early ----
#pragma unroll
    for (int t = 0; t < 8; ++t) {
      const int keyl = t * 4 + kl4;
      const int key = kb + keyl;
      bf16x8 vv = {};
      if (key < limit_k)
        vv = *(const bf16x8*)(vb + (size_t)key * QKV_DIM + c8);
#pragma unroll
      for (int j = 0; j < 8; ++j) sv[(c8 + j) * PSTR + keyl] = vv[j];
    }

    // ---- scores: two 16x16 key tiles, K=hd=64 -> 2 WMMAs each ----
    v8f S0 = {}, S1 = {};
    S0 = wmma_bf16(qa0, load_kfrag(kb_, kb, 0, QKV_DIM, limit_k), S0);
    S0 = wmma_bf16(qa1, load_kfrag(kb_, kb, 32, QKV_DIM, limit_k), S0);
    S1 = wmma_bf16(qa0, load_kfrag(kb_, kb + 16, 0, QKV_DIM, limit_k), S1);
    S1 = wmma_bf16(qa1, load_kfrag(kb_, kb + 16, 32, QKV_DIM, limit_k), S1);

    const bool v0 = (kb + col) < limit_k;
    const bool v1 = (kb + 16 + col) < limit_k;

    // ---- online softmax (max via permlane butterflies; sums via WMMA) ----
#pragma unroll
    for (int r = 0; r < 8; ++r) {
      float s0 = v0 ? S0[r] * scale : -1e30f;
      float s1 = v1 ? S1[r] * scale : -1e30f;
      const float tmax = rowmax16(fmaxf(s0, s1));
      const float mnew = fmaxf(m[r], tmax);
      const float corr = __expf(m[r] - mnew);
      m[r] = mnew;
      l[r] *= corr;
#pragma unroll
      for (int j = 0; j < 4; ++j) acc[j][r] *= corr;
      sp[(r + rh) * PSTR + col] = to_bf16(__expf(s0 - mnew));
      sp[(r + rh) * PSTR + 16 + col] = to_bf16(__expf(s1 - mnew));
    }
    __builtin_amdgcn_wave_barrier();

    // ---- rebuild P as A-fragment (two b128 LDS reads, already bf16) ----
    const __bf16* pp = sp + col * PSTR + ((lane >> 4) << 4);
    bf16x16 pa = cat8(*(const bf16x8*)pp, *(const bf16x8*)(pp + 8));

    // ---- row sums: P(16x32) x ones(32x16) -> every column = row sum ----
    {
      v8f z = {};
      v8f Ssum = wmma_bf16(pa, ones, z);
#pragma unroll
      for (int r = 0; r < 8; ++r) l[r] += Ssum[r];
    }

    // ---- O += P(16x32) * V(32x64): 4 WMMAs, B-frags from LDS ----
#pragma unroll
    for (int j = 0; j < 4; ++j) {
      const __bf16* vp = sv + (j * 16 + col) * PSTR + ((lane >> 4) << 4);
      bf16x16 vf = cat8(*(const bf16x8*)vp, *(const bf16x8*)(vp + 8));
      acc[j] = wmma_bf16(pa, vf, acc[j]);
    }
    __builtin_amdgcn_wave_barrier();
  }

  // ---- epilogue: normalize and store bf16 (guard partial text tile) ----
#pragma unroll
  for (int r = 0; r < 8; ++r) {
    const int qrow = q0 + r + rh;
    if (qrow < limit_q) {
      const float inv = 1.0f / l[r];
#pragma unroll
      for (int j = 0; j < 4; ++j) {
        out[(tokbase + qrow) * (size_t)C_DIM + h * 64 + j * 16 + col] =
            to_bf16(acc[j][r] * inv);
      }
    }
  }
}

// ---------------------------------------------------------------------------
// Host launcher. Workspace layout (bytes, all 256-aligned):
//   [0)          xb      : 8512*768  bf16  = 13,074,432
//   [13074432)   qkv_wb  : 2304*768  bf16  =  3,538,944
//   [16613376)   proj_wb : 768*768   bf16  =  1,179,648
//   [17793024)   qkvb    : 8512*2304 bf16  = 39,223,296
//   [57016320)   attn_o  : 8512*768  bf16  = 13,074,432   (total ~70.1 MB)
// ---------------------------------------------------------------------------
extern "C" void kernel_launch(void* const* d_in, const int* in_sizes, int n_in,
                              void* d_out, int out_size, void* d_ws,
                              size_t ws_size, hipStream_t stream) {
  (void)in_sizes; (void)n_in; (void)out_size; (void)ws_size;
  const float* x = (const float*)d_in[0];
  const float* qkv_w = (const float*)d_in[1];
  const float* proj_w = (const float*)d_in[2];
  const float* proj_b = (const float*)d_in[3];
  float* out = (float*)d_out;

  char* ws = (char*)d_ws;
  __bf16* xb = (__bf16*)(ws);
  __bf16* qwb = (__bf16*)(ws + 13074432);
  __bf16* pwb = (__bf16*)(ws + 16613376);
  __bf16* qkvb = (__bf16*)(ws + 17793024);
  __bf16* ao = (__bf16*)(ws + 57016320);

  const int nx = M_TOK * C_DIM;        // 6,537,216
  const int nqw = QKV_DIM * C_DIM;     // 1,769,472
  const int npw = C_DIM * C_DIM;       // 589,824
  f32_to_bf16<<<(nx + 255) / 256, 256, 0, stream>>>(x, xb, nx);
  f32_to_bf16<<<(nqw + 255) / 256, 256, 0, stream>>>(qkv_w, qwb, nqw);
  f32_to_bf16<<<(npw + 255) / 256, 256, 0, stream>>>(proj_w, pwb, npw);

  // QKV projection: M=8512, N=2304, K=768; 34 M-blocks x 36 N-groups
  gemm_wmma<true><<<34 * 36, 256, 0, stream>>>(xb, qwb, qkvb, nullptr, nullptr,
                                               M_TOK, QKV_DIM, C_DIM);

  // Attention: 8 batches * 12 heads * 67 query tiles = 6432 waves, 4/block
  attn_wmma<<<6432 / ATTN_WAVES, 32 * ATTN_WAVES, 0, stream>>>(qkvb, ao);

  // Output projection: M=8512, N=768, K=768; 34 M-blocks x 12 N-groups
  gemm_wmma<false><<<34 * 12, 256, 0, stream>>>(ao, pwb, nullptr, out, proj_b,
                                                M_TOK, C_DIM, C_DIM);
}
